// FALR7_35794257445099
// MI455X (gfx1250) — matrix operand order repaired and checked by reading, not compiled, not run on hardware
//
#include <hip/hip_runtime.h>

// B=8, N=256, D=E=G=MID=OUT=128
#define Bc 8
#define Nc 256
#define Kc 128
#define JT 32            // j-tile width (2 WMMA M-tiles per wave)
#define NJT (Nc / JT)    // 8
#define NCHUNK 4         // i-range split for parallelism
#define ICH (Nc / NCHUNK)

typedef _Float16 h16 __attribute__((ext_vector_type(16)));
typedef _Float16 h8  __attribute__((ext_vector_type(8)));
typedef float    f8  __attribute__((ext_vector_type(8)));

// ---------------------------------------------------------------------------
// Kernel 1: prep. Per output element (b,n,c):
//   a_buf    = node@Wn + bn + be + (graph@Wg + bg)   (receiver term, biases folded)
//   h_buf    = hidden@Wh + bh                        (sender term)
//   base_out = node@Wo1 + hidden@Wo2 + bo1+bo2+bo3   (output skip term)
// ---------------------------------------------------------------------------
__global__ __launch_bounds__(256)
void FALR7_prep(const float* __restrict__ node, const float* __restrict__ hidden,
                const float* __restrict__ graph,
                const float* __restrict__ Wn, const float* __restrict__ bn,
                const float* __restrict__ Wh, const float* __restrict__ bh,
                const float* __restrict__ be,
                const float* __restrict__ Wg, const float* __restrict__ bg,
                const float* __restrict__ Wo1, const float* __restrict__ bo1,
                const float* __restrict__ Wo2, const float* __restrict__ bo2,
                const float* __restrict__ bo3,
                float* __restrict__ a_buf, float* __restrict__ h_buf,
                float* __restrict__ base_out)
{
    const int idx  = blockIdx.x * 256 + threadIdx.x;   // (b*N+n)*K + c
    const int c    = idx & (Kc - 1);
    const int rowg = idx >> 7;                          // b*N + n
    const int b    = rowg >> 8;                         // N == 256

    const float* nf = node   + (size_t)rowg * Kc;
    const float* hf = hidden + (size_t)rowg * Kc;
    const float* gf = graph  + (size_t)b    * Kc;

    float accA = bn[c] + be[c] + bg[c];
    float accH = bh[c];
    float accO = bo1[c] + bo2[c] + bo3[c];
    #pragma unroll 4
    for (int k = 0; k < Kc; ++k) {
        const float nk = nf[k], hk = hf[k], gk = gf[k];
        accA = fmaf(nk, Wn [k * Kc + c], accA);
        accA = fmaf(gk, Wg [k * Kc + c], accA);
        accH = fmaf(hk, Wh [k * Kc + c], accH);
        accO = fmaf(nk, Wo1[k * Kc + c], accO);
        accO = fmaf(hk, Wo2[k * Kc + c], accO);
    }
    a_buf[idx]    = accA;
    h_buf[idx]    = accH;
    base_out[idx] = accO;
}

// ---------------------------------------------------------------------------
// Kernel 2: fused edge GEMM + broadcast-add + masked max over senders i.
// Workgroup = (b, 32-wide j-tile, 64-wide i-chunk). 8 wave32s; wave w owns MID
// columns [16w,16w+16) and runs two independent WMMA chains (j rows 0-15 and
// 16-31 of the tile). Double-buffered LDS tile -> one barrier per i-step.
// partial[b,chunk,j,c] = max_{i in chunk} adj(b,i,j)*(msg_e + h[i] + a[j])
// ---------------------------------------------------------------------------
__global__ __launch_bounds__(256)
void FALR7_msgmax(const float* __restrict__ edge, const int* __restrict__ adj,
                  const float* __restrict__ We,
                  const float* __restrict__ a_buf, const float* __restrict__ h_buf,
                  float* __restrict__ partial)
{
    __shared__ alignas(16) _Float16 As[2][JT][Kc + 8];  // ping-pong, +16B row pad

    const int tid  = threadIdx.x;
    const int wave = tid >> 5, lane = tid & 31;
    const int lo   = lane & 15, hi = lane >> 4;
    const int c0   = wave * 16;

    int wg = blockIdx.x;
    const int chunk = wg % NCHUNK;  wg /= NCHUNK;
    const int jt    = wg % NJT;     wg /= NJT;
    const int b     = wg;
    const int j0    = jt * JT;

    // B fragments of We (32x16 f16 layout: element e holds K=32*kk+8*hi+(e&7)+16*(e>>3),
    // column c0+lo). Resident in VGPRs for the whole chunk.
    h16 wb[4];
    #pragma unroll
    for (int kk = 0; kk < 4; ++kk)
        #pragma unroll
        for (int e = 0; e < 16; ++e)
            wb[kk][e] = (_Float16)We[(kk * 32 + hi * 8 + (e & 7) + (e >> 3) * 16) * Kc + (c0 + lo)];

    // Receiver terms for both j sub-tiles (row m = 8*hi + r matches C layout).
    float av0[8], av1[8];
    #pragma unroll
    for (int r = 0; r < 8; ++r) {
        av0[r] = a_buf[(size_t)(b * Nc + j0 +      hi * 8 + r) * Kc + (c0 + lo)];
        av1[r] = a_buf[(size_t)(b * Nc + j0 + 16 + hi * 8 + r) * Kc + (c0 + lo)];
    }

    f8 mx0, mx1;
    #pragma unroll
    for (int r = 0; r < 8; ++r) { mx0[r] = -__builtin_inff(); mx1[r] = -__builtin_inff(); }

    const int row = tid >> 4;          // 0..15 -> handles tile rows row, row+16
    const int k0  = (tid & 15) * 8;    // 8-float chunk of K

    auto tileAddr = [&](int i, int r) -> const float4* {
        return (const float4*)(edge + ((size_t)((b * Nc + i) * Nc + (j0 + r))) * Kc + k0);
    };
    auto cvtStore = [&](int pbuf, float4 q0, float4 q1, int r) {
        h8 v;
        v[0] = (_Float16)q0.x; v[1] = (_Float16)q0.y;
        v[2] = (_Float16)q0.z; v[3] = (_Float16)q0.w;
        v[4] = (_Float16)q1.x; v[5] = (_Float16)q1.y;
        v[6] = (_Float16)q1.z; v[7] = (_Float16)q1.w;
        *(h8*)&As[pbuf][r][k0] = v;
    };

    const int i0 = chunk * ICH;

    // Prologue: stage tile for ii=0 into buffer 0.
    float4 p0, p1, p2, p3;
    { const float4* s0 = tileAddr(i0, row);      p0 = s0[0]; p1 = s0[1];
      const float4* s1 = tileAddr(i0, row + 16); p2 = s1[0]; p3 = s1[1]; }
    cvtStore(0, p0, p1, row);
    cvtStore(0, p2, p3, row + 16);
    __syncthreads();

    for (int ii = 0; ii < ICH; ++ii) {
        const int i = i0 + ii;
        const int p = ii & 1;

        // Per-i scalars: sender term broadcast + adjacency row (no divergence,
        // two aligned int4 loads per sub-tile; whole block shares one 128B row).
        const float hv = h_buf[(size_t)(b * Nc + i) * Kc + (c0 + lo)];
        const int* ar = adj + (size_t)(b * Nc + i) * Nc + j0;
        const int4 q0 = *(const int4*)(ar + hi * 8);
        const int4 q1 = *(const int4*)(ar + hi * 8 + 4);
        const int4 q2 = *(const int4*)(ar + 16 + hi * 8);
        const int4 q3 = *(const int4*)(ar + 16 + hi * 8 + 4);

        // Software pipeline: issue next tile's global loads before compute.
        if (ii + 1 < ICH) {
            const float4* s0 = tileAddr(i + 1, row);      p0 = s0[0]; p1 = s0[1];
            const float4* s1 = tileAddr(i + 1, row + 16); p2 = s1[0]; p3 = s1[1];
            if (ii + 2 < ICH)
                __builtin_prefetch(tileAddr(i + 2, row), 0, 1);  // global_prefetch_b8
        }

        // Two interleaved WMMA chains from buffer p.
        const _Float16* base0 = &As[p][lo][0];
        const _Float16* base1 = &As[p][lo + 16][0];
        f8 cacc0 = {}, cacc1 = {};
        #pragma unroll
        for (int kk = 0; kk < 4; ++kk) {
            const int kb = kk * 32 + hi * 8;
            const h8 x0 = *(const h8*)(base0 + kb);
            const h8 x1 = *(const h8*)(base0 + kb + 16);
            const h8 y0 = *(const h8*)(base1 + kb);
            const h8 y1 = *(const h8*)(base1 + kb + 16);
            const h16 af0 = __builtin_shufflevector(x0, x1,
                0, 1, 2, 3, 4, 5, 6, 7, 8, 9, 10, 11, 12, 13, 14, 15);
            const h16 af1 = __builtin_shufflevector(y0, y1,
                0, 1, 2, 3, 4, 5, 6, 7, 8, 9, 10, 11, 12, 13, 14, 15);
            cacc0 = __builtin_amdgcn_wmma_f32_16x16x32_f16(
                false, af0, false, wb[kk], (short)0, cacc0, false, false);
            cacc1 = __builtin_amdgcn_wmma_f32_16x16x32_f16(
                false, af1, false, wb[kk], (short)0, cacc1, false, false);
        }

        // Stage next tile into the other buffer (no barrier needed in between:
        // writes go to 1-p while reads came from p).
        if (ii + 1 < ICH) {
            cvtStore(1 - p, p0, p1, row);
            cvtStore(1 - p, p2, p3, row + 16);
        }

        // Epilogue: add sender/receiver terms, mask, running max.
        const float am0[8] = {(float)q0.x, (float)q0.y, (float)q0.z, (float)q0.w,
                              (float)q1.x, (float)q1.y, (float)q1.z, (float)q1.w};
        const float am1[8] = {(float)q2.x, (float)q2.y, (float)q2.z, (float)q2.w,
                              (float)q3.x, (float)q3.y, (float)q3.z, (float)q3.w};
        #pragma unroll
        for (int r = 0; r < 8; ++r) {
            mx0[r] = fmaxf(mx0[r], am0[r] * (cacc0[r] + hv + av0[r]));
            mx1[r] = fmaxf(mx1[r], am1[r] * (cacc1[r] + hv + av1[r]));
        }

        __syncthreads();
    }

    #pragma unroll
    for (int r = 0; r < 8; ++r) {
        const size_t base = (size_t)(b * NCHUNK + chunk) * Nc;
        partial[(base + (j0 +      hi * 8 + r)) * Kc + (c0 + lo)] = mx0[r];
        partial[(base + (j0 + 16 + hi * 8 + r)) * Kc + (c0 + lo)] = mx1[r];
    }
}

// ---------------------------------------------------------------------------
// Kernel 3: chunk-max reduce -> msgs tile (f16 in LDS) -> WMMA with Wo3 ->
// add base_out -> d_out. One workgroup per 16 rows of (B*N).
// ---------------------------------------------------------------------------
__global__ __launch_bounds__(256)
void FALR7_out(const float* __restrict__ partial, const float* __restrict__ Wo3,
               const float* __restrict__ base_out, float* __restrict__ out)
{
    __shared__ alignas(16) _Float16 Ms[16][Kc + 8];

    const int tid  = threadIdx.x;
    const int wave = tid >> 5, lane = tid & 31;
    const int lo   = lane & 15, hi = lane >> 4;
    const int c0   = wave * 16;
    const int row  = tid >> 4;
    const int k0   = (tid & 15) * 8;
    const int gr0  = blockIdx.x * 16;   // row base in [0, B*N)

    {   // max over NCHUNK partials, convert to f16 tile
        const int gr = gr0 + row;
        const int b  = gr >> 8, j = gr & 255;
        float m[8];
        #pragma unroll
        for (int e = 0; e < 8; ++e) m[e] = -__builtin_inff();
        #pragma unroll
        for (int ch = 0; ch < NCHUNK; ++ch) {
            const float* p = partial + ((size_t)(b * NCHUNK + ch) * Nc + j) * Kc + k0;
            const float4 p0 = *(const float4*)p;
            const float4 p1 = *(const float4*)(p + 4);
            m[0] = fmaxf(m[0], p0.x); m[1] = fmaxf(m[1], p0.y);
            m[2] = fmaxf(m[2], p0.z); m[3] = fmaxf(m[3], p0.w);
            m[4] = fmaxf(m[4], p1.x); m[5] = fmaxf(m[5], p1.y);
            m[6] = fmaxf(m[6], p1.z); m[7] = fmaxf(m[7], p1.w);
        }
        h8 v;
        #pragma unroll
        for (int e = 0; e < 8; ++e) v[e] = (_Float16)m[e];
        *(h8*)&Ms[row][k0] = v;
    }
    __syncthreads();

    h16 wb[4];
    #pragma unroll
    for (int kk = 0; kk < 4; ++kk)
        #pragma unroll
        for (int e = 0; e < 16; ++e)
            wb[kk][e] = (_Float16)Wo3[(kk * 32 + hi * 8 + (e & 7) + (e >> 3) * 16) * Kc + (c0 + lo)];

    f8 cacc = {};
    #pragma unroll
    for (int kk = 0; kk < 4; ++kk) {
        const int kb = kk * 32 + hi * 8;
        const h8 a0 = *(const h8*)&Ms[lo][kb];
        const h8 a1 = *(const h8*)&Ms[lo][kb + 16];
        const h16 af = __builtin_shufflevector(a0, a1,
            0, 1, 2, 3, 4, 5, 6, 7, 8, 9, 10, 11, 12, 13, 14, 15);
        cacc = __builtin_amdgcn_wmma_f32_16x16x32_f16(
            false, af, false, wb[kk], (short)0, cacc, false, false);
    }

    #pragma unroll
    for (int r = 0; r < 8; ++r) {
        const int gr = gr0 + hi * 8 + r;
        const size_t o = (size_t)gr * Kc + (c0 + lo);
        out[o] = cacc[r] + base_out[o];
    }
}

// ---------------------------------------------------------------------------
extern "C" void kernel_launch(void* const* d_in, const int* in_sizes, int n_in,
                              void* d_out, int out_size, void* d_ws, size_t ws_size,
                              hipStream_t stream) {
    const float* node   = (const float*)d_in[0];
    const float* edge   = (const float*)d_in[1];
    const float* graph  = (const float*)d_in[2];
    const int*   adj    = (const int*)  d_in[3];
    const float* hidden = (const float*)d_in[4];
    const float* Wn  = (const float*)d_in[5];  const float* bn  = (const float*)d_in[6];
    const float* Wh  = (const float*)d_in[7];  const float* bh  = (const float*)d_in[8];
    const float* We  = (const float*)d_in[9];  const float* be  = (const float*)d_in[10];
    const float* Wg  = (const float*)d_in[11]; const float* bg  = (const float*)d_in[12];
    const float* Wo1 = (const float*)d_in[13]; const float* bo1 = (const float*)d_in[14];
    const float* Wo2 = (const float*)d_in[15]; const float* bo2 = (const float*)d_in[16];
    const float* Wo3 = (const float*)d_in[17]; const float* bo3 = (const float*)d_in[18];
    float* out = (float*)d_out;

    char* ws = (char*)d_ws;
    const size_t rowsz = (size_t)Bc * Nc * Kc * sizeof(float);   // 1 MiB
    float* a_buf    = (float*)(ws);
    float* h_buf    = (float*)(ws + rowsz);
    float* base_out = (float*)(ws + 2 * rowsz);
    float* partial  = (float*)(ws + 3 * rowsz);                  // NCHUNK MiB

    FALR7_prep<<<(Bc * Nc * Kc) / 256, 256, 0, stream>>>(
        node, hidden, graph, Wn, bn, Wh, bh, be, Wg, bg,
        Wo1, bo1, Wo2, bo2, bo3, a_buf, h_buf, base_out);

    FALR7_msgmax<<<Bc * NJT * NCHUNK, 256, 0, stream>>>(
        edge, adj, We, a_buf, h_buf, partial);

    FALR7_out<<<(Bc * Nc) / 16, 256, 0, stream>>>(
        partial, Wo3, base_out, out);
}
